// BasicBlock_81166291960009
// MI455X (gfx1250) — matrix operand order repaired and checked by reading, not compiled, not run on hardware
//
#include <hip/hip_runtime.h>
#include <hip/hip_bf16.h>

typedef __attribute__((ext_vector_type(16))) _Float16 v16h;
typedef __attribute__((ext_vector_type(8)))  _Float16 v8h;
typedef __attribute__((ext_vector_type(8)))  float    v8f;

#define NN 8
#define CC 64
#define HH 128
#define WW 128
#define KTOT 576              // 64 cin * 9 taps
#define LX_COLS 132           // 130 used (+halo), padded
#define LW_HALVES (CC * KTOT)            // 36864 halves (72 KB)
#define LX_HALVES (3 * LX_COLS * CC)     // 25344 halves (~49.5 KB)
#define SMEM_BYTES ((LW_HALVES + LX_HALVES) * 2 + 2 * CC * 4)

// ---------------------------------------------------------------------------
// Kernel 1: both 64->1 mask convs, sigmoid, hard threshold (shared x loads)
// ---------------------------------------------------------------------------
__global__ __launch_bounds__(256) void mask_logits_kernel(
    const float* __restrict__ x,
    const float* __restrict__ mw1, const float* __restrict__ mb1,
    const float* __restrict__ mw2, const float* __restrict__ mb2,
    float* __restrict__ m1, float* __restrict__ m2)
{
    __shared__ float w1[KTOT];
    __shared__ float w2[KTOT];
    for (int i = threadIdx.x; i < KTOT; i += 256) { w1[i] = mw1[i]; w2[i] = mw2[i]; }
    __syncthreads();

    int idx = blockIdx.x * 256 + threadIdx.x;           // [0, N*H*W)
    int w = idx % WW;
    int h = (idx / WW) % HH;
    int n = idx / (WW * HH);

    float l1 = mb1[0], l2 = mb2[0];
    for (int cin = 0; cin < CC; ++cin) {
        const float* xp = x + ((size_t)(n * CC + cin) * HH) * WW;
        const float* wp1 = w1 + cin * 9;
        const float* wp2 = w2 + cin * 9;
        #pragma unroll
        for (int kh = 0; kh < 3; ++kh) {
            int hh = h + kh - 1;
            if (hh < 0 || hh >= HH) continue;
            #pragma unroll
            for (int kw = 0; kw < 3; ++kw) {
                int ww = w + kw - 1;
                if (ww < 0 || ww >= WW) continue;
                float xv = xp[hh * WW + ww];
                int t = kh * 3 + kw;
                l1 = fmaf(wp1[t], xv, l1);
                l2 = fmaf(wp2[t], xv, l2);
            }
        }
    }
    float p1 = 1.0f / (1.0f + expf(-l1));
    float p2 = 1.0f / (1.0f + expf(-l2));
    m1[idx] = (p1 >= 0.5f) ? p1 : 0.0f;
    m2[idx] = (p2 >= 0.5f) ? p2 : 0.0f;
}

// ---------------------------------------------------------------------------
// Kernel 2: 3x3 normalized Gaussian smoothing of a hard mask (zero padded)
// ---------------------------------------------------------------------------
__global__ __launch_bounds__(256) void gauss_smooth_kernel(
    const float* __restrict__ m, const float* __restrict__ sigma,
    float* __restrict__ nm)
{
    int idx = blockIdx.x * 256 + threadIdx.x;
    int w = idx % WW;
    int h = (idx / WW) % HH;
    int n = idx / (WW * HH);

    float s = sigma[0];
    float e = expf(-1.0f / (2.0f * s * s));  // weight for |d| == 1
    float g[3] = { e, 1.0f, e };
    float gs = g[0] + g[1] + g[2];
    float norm = 1.0f / (gs * gs);

    const float* mp = m + (size_t)n * HH * WW;
    float acc = 0.0f;
    #pragma unroll
    for (int j = 0; j < 3; ++j) {
        int hh = h + j - 1;
        if (hh < 0 || hh >= HH) continue;
        #pragma unroll
        for (int i = 0; i < 3; ++i) {
            int ww = w + i - 1;
            if (ww < 0 || ww >= WW) continue;
            acc = fmaf(g[j] * g[i], mp[hh * WW + ww], acc);
        }
    }
    nm[idx] = acc * norm;
}

// ---------------------------------------------------------------------------
// Kernel 3: 3x3 64->64 implicit-GEMM conv via v_wmma_f32_16x16x32_f16,
// fused BN + mask (+ residual + relu). One workgroup per (n,h) row.
//   GEMM: M = cout (64), K = tap*64 + cin (576), N = 128 pixels of the row.
//   8 waves: wave w -> cout-group (w&3), pixel-groups ((w>>2)*4 .. +3).
// Inner loop: group all fragment loads before the 4 WMMAs so the scheduler
// can batch ds_load_b128s, use partial dscnt waits, and overlap the next
// chunk's loads with the current chunk's WMMAs (no operand-overlap hazards).
// ---------------------------------------------------------------------------
__global__ __launch_bounds__(256) void conv3x3_wmma_kernel(
    const float* __restrict__ xin,   // activations [N,C,H,W]
    const float* __restrict__ wgt,   // weights [C,C,3,3]
    const float* __restrict__ gam, const float* __restrict__ bet,
    const float* __restrict__ mu,  const float* __restrict__ var,
    const float* __restrict__ nm,    // smoothed mask [N,H,W]
    const float* __restrict__ resid, // residual (x) for second conv
    float* __restrict__ dst,
    int second)
{
    extern __shared__ char smem[];
    _Float16* lw = (_Float16*)smem;              // [cout][K]  K = tap*64+cin
    _Float16* lx = lw + LW_HALVES;               // [row(3)][col(132)][cin(64)]
    float*    sc = (float*)(lx + LX_HALVES);     // BN scale[64]
    float*    sh = sc + CC;                      // BN shift[64]

    const int tid = threadIdx.x;
    const int n = blockIdx.x / HH;
    const int h = blockIdx.x % HH;

    // Stage weights f32 -> f16 (source-major: contiguous global reads).
    // Global layout j = (cout*64 + cin)*9 + tap ; LDS dest = cout*576 + tap*64 + cin
    for (int j = tid; j < LW_HALVES; j += 256) {
        int tap    = j % 9;
        int cincout= j / 9;          // cout*64 + cin
        int cin    = cincout & 63;
        int cout   = cincout >> 6;
        lw[cout * KTOT + tap * CC + cin] = (_Float16)wgt[j];
    }
    // Stage 3-row activation slab with column halo, cin-contiguous in LDS.
    for (int rr = 0; rr < 3; ++rr) {
        int hh = h - 1 + rr;
        for (int i = tid; i < 130 * CC; i += 256) {
            int cin = i / 130;
            int c   = i % 130;       // c -> global col w = c-1
            int w   = c - 1;
            float v = 0.0f;
            if (hh >= 0 && hh < HH && w >= 0 && w < WW)
                v = xin[((n * CC + cin) * HH + hh) * WW + w];
            lx[(rr * LX_COLS + c) * CC + cin] = (_Float16)v;
        }
    }
    if (tid < CC) {
        float s = gam[tid] * rsqrtf(var[tid] + 1e-5f);
        sc[tid] = s;
        sh[tid] = bet[tid] - mu[tid] * s;
    }
    __syncthreads();

    const int wave = tid >> 5;
    const int lane = tid & 31;
    const int cg   = wave & 3;          // cout group (16 couts)
    const int pgb  = (wave >> 2) * 4;   // first pixel group (16 pixels each)
    const int lgrp = lane >> 4;         // 0/1 half-wave
    const int lm   = lane & 15;

    // Per-lane fragment base addresses, hoisted out of the K-loop so all
    // per-chunk deltas fold into ds_load immediate offsets.
    const _Float16* abase = lw + (cg * 16 + lm) * KTOT + lgrp * 16;
    const _Float16* bbase0 = lx + ((pgb + 0) * 16 + lm) * CC + lgrp * 16;
    const _Float16* bbase1 = lx + ((pgb + 1) * 16 + lm) * CC + lgrp * 16;
    const _Float16* bbase2 = lx + ((pgb + 2) * 16 + lm) * CC + lgrp * 16;
    const _Float16* bbase3 = lx + ((pgb + 3) * 16 + lm) * CC + lgrp * 16;

    v8f acc0 = (v8f){0,0,0,0,0,0,0,0};
    v8f acc1 = (v8f){0,0,0,0,0,0,0,0};
    v8f acc2 = (v8f){0,0,0,0,0,0,0,0};
    v8f acc3 = (v8f){0,0,0,0,0,0,0,0};

    #pragma unroll
    for (int kc = 0; kc < 18; ++kc) {   // K chunks of 32: tap = kc>>1, cin half
        const int tap  = kc >> 1;
        const int kh   = tap / 3;
        const int kw   = tap % 3;
        const int coff = (kh * LX_COLS + kw) * CC + (kc & 1) * 32;

        // --- issue ALL fragment loads for this chunk first ---
        const _Float16* ap = abase + kc * 32;
        v8h alo = *(const v8h*)ap;          // K = kb0..kb0+7
        v8h ahi = *(const v8h*)(ap + 8);    // K = kb0+8..kb0+15

        const _Float16* bp0 = bbase0 + coff;
        const _Float16* bp1 = bbase1 + coff;
        const _Float16* bp2 = bbase2 + coff;
        const _Float16* bp3 = bbase3 + coff;
        v8h b0lo = *(const v8h*)bp0; v8h b0hi = *(const v8h*)(bp0 + 8);
        v8h b1lo = *(const v8h*)bp1; v8h b1hi = *(const v8h*)(bp1 + 8);
        v8h b2lo = *(const v8h*)bp2; v8h b2hi = *(const v8h*)(bp2 + 8);
        v8h b3lo = *(const v8h*)bp3; v8h b3hi = *(const v8h*)(bp3 + 8);

        v16h a  = __builtin_shufflevector(alo,  ahi,  0,1,2,3,4,5,6,7,8,9,10,11,12,13,14,15);
        v16h b0 = __builtin_shufflevector(b0lo, b0hi, 0,1,2,3,4,5,6,7,8,9,10,11,12,13,14,15);
        v16h b1 = __builtin_shufflevector(b1lo, b1hi, 0,1,2,3,4,5,6,7,8,9,10,11,12,13,14,15);
        v16h b2 = __builtin_shufflevector(b2lo, b2hi, 0,1,2,3,4,5,6,7,8,9,10,11,12,13,14,15);
        v16h b3 = __builtin_shufflevector(b3lo, b3hi, 0,1,2,3,4,5,6,7,8,9,10,11,12,13,14,15);

        // --- then the 4 WMMAs (distinct D/A/B regs: no hazard NOPs needed) ---
        acc0 = __builtin_amdgcn_wmma_f32_16x16x32_f16(false, a, false, b0, (short)0, acc0, false, false);
        acc1 = __builtin_amdgcn_wmma_f32_16x16x32_f16(false, a, false, b1, (short)0, acc1, false, false);
        acc2 = __builtin_amdgcn_wmma_f32_16x16x32_f16(false, a, false, b2, (short)0, acc2, false, false);
        acc3 = __builtin_amdgcn_wmma_f32_16x16x32_f16(false, a, false, b3, (short)0, acc3, false, false);
    }

    // Epilogue: C/D layout -> VGPR r: M = r + lgrp*8, N = lm.
    // Hoist BN scale/shift for this wave's 8 couts into registers.
    float lscale[8], lshift[8];
    #pragma unroll
    for (int r = 0; r < 8; ++r) {
        int cout = cg * 16 + lgrp * 8 + r;
        lscale[r] = sc[cout];
        lshift[r] = sh[cout];
    }

    v8f accs[4] = { acc0, acc1, acc2, acc3 };
    #pragma unroll
    for (int pgi = 0; pgi < 4; ++pgi) {
        int pix = (pgb + pgi) * 16 + lm;
        float nmv = nm[(n * HH + h) * WW + pix];
        #pragma unroll
        for (int r = 0; r < 8; ++r) {
            int cout = cg * 16 + lgrp * 8 + r;
            float v = accs[pgi][r] * lscale[r] + lshift[r];
            int gi = ((n * CC + cout) * HH + h) * WW + pix;
            if (!second) {
                v = fmaxf(v, 0.0f) * nmv;           // relu(bn(conv1)) * nm1
            } else {
                v = fmaf(v, nmv, resid[gi]);        // bn(conv2)*nm2 + residual
                v = fmaxf(v, 0.0f);
            }
            dst[gi] = v;
        }
    }
}

// ---------------------------------------------------------------------------
extern "C" void kernel_launch(void* const* d_in, const int* in_sizes, int n_in,
                              void* d_out, int out_size, void* d_ws, size_t ws_size,
                              hipStream_t stream) {
    (void)in_sizes; (void)n_in; (void)out_size; (void)ws_size;

    const float* x    = (const float*)d_in[0];
    const float* c1w  = (const float*)d_in[1];
    const float* c2w  = (const float*)d_in[2];
    const float* m1w  = (const float*)d_in[3];
    const float* m1b  = (const float*)d_in[4];
    const float* m2w  = (const float*)d_in[5];
    const float* m2b  = (const float*)d_in[6];
    const float* g1   = (const float*)d_in[7];
    const float* b1   = (const float*)d_in[8];
    const float* mu1  = (const float*)d_in[9];
    const float* v1   = (const float*)d_in[10];
    const float* g2   = (const float*)d_in[11];
    const float* b2   = (const float*)d_in[12];
    const float* mu2  = (const float*)d_in[13];
    const float* v2   = (const float*)d_in[14];
    const float* sig1 = (const float*)d_in[15];
    const float* sig2 = (const float*)d_in[16];

    float* out = (float*)d_out;
    float* ws  = (float*)d_ws;

    const int npix = NN * HH * WW;                 // 131072
    float* out1 = ws;                              // N*C*H*W floats
    float* m1   = out1 + (size_t)NN * CC * HH * WW;
    float* m2   = m1 + npix;
    float* nm1  = m2 + npix;
    float* nm2  = nm1 + npix;

    mask_logits_kernel<<<npix / 256, 256, 0, stream>>>(x, m1w, m1b, m2w, m2b, m1, m2);
    gauss_smooth_kernel<<<npix / 256, 256, 0, stream>>>(m1, sig1, nm1);
    gauss_smooth_kernel<<<npix / 256, 256, 0, stream>>>(m2, sig2, nm2);

    conv3x3_wmma_kernel<<<NN * HH, 256, SMEM_BYTES, stream>>>(
        x, c1w, g1, b1, mu1, v1, nm1, x, out1, 0);
    conv3x3_wmma_kernel<<<NN * HH, 256, SMEM_BYTES, stream>>>(
        out1, c2w, g2, b2, mu2, v2, nm2, x, out, 1);
}
